// BaseRNN_20091857011133
// MI455X (gfx1250) — matrix operand order repaired and checked by reading, not compiled, not run on hardware
//
#include <hip/hip_runtime.h>
#include <hip/hip_bf16.h>
#include <math.h>

typedef __attribute__((ext_vector_type(16))) _Float16 v16h;
typedef __attribute__((ext_vector_type(8)))  _Float16 v8h;
typedef __attribute__((ext_vector_type(8)))  float    v8f;

#define B_ 16
#define T_ 4096
#define D_ 128
#define H_ 128
#define G_ 512   // 4*H

// ---- LDS layout for the scan kernel (byte offsets) ----
#define LDS_U_BYTES    (D_ * G_ * 2)            // 131072
#define LDS_H_BYTES    (2 * 16 * H_ * 2)        // 8192
#define LDS_XW_TILE    (B_ * G_ * 2)            // 16384 per buffer
#define LDS_XW_OFF     (LDS_U_BYTES + LDS_H_BYTES)
#define LDS_TOTAL      (LDS_XW_OFF + 2 * LDS_XW_TILE)   // 172032

__device__ __forceinline__ float fast_sigmoid(float x) {
    return 1.0f / (1.0f + __expf(-x));
}
__device__ __forceinline__ float fast_tanh(float x) {
    return 2.0f / (1.0f + __expf(-2.0f * x)) - 1.0f;
}

// ---------------------------------------------------------------------------
// Tensor Data Mover: 2D f16 tile (rows x row_len, densely packed) -> LDS.
// D# per CDNA5 ISA; 6-arg builtin on this toolchain.
// ---------------------------------------------------------------------------
#if __has_builtin(__builtin_amdgcn_tensor_load_to_lds)
#define HAVE_TDM 1
typedef __attribute__((ext_vector_type(4))) unsigned int v4u;
typedef __attribute__((ext_vector_type(8))) int          v8i_;
typedef __attribute__((ext_vector_type(4))) int          v4i_;

__device__ __forceinline__ void tdm_load_2d_f16(unsigned lds_byte_off,
                                                const void* gaddr,
                                                unsigned rows,
                                                unsigned row_len) {
    unsigned long long ga = (unsigned long long)(size_t)gaddr;
    v4u g0;
    g0[0] = 1u;                                        // count=1 (valid user D#)
    g0[1] = lds_byte_off;                              // lds_addr
    g0[2] = (unsigned)(ga & 0xffffffffu);              // global_addr[31:0]
    g0[3] = (unsigned)((ga >> 32) & 0x1ffffffu)        // global_addr[56:32]
          | (2u << 30);                                // type=2 ("image")
    v8i_ g1;
    g1[0] = 0x00010000;                                // data_size=1 (2 bytes)
    g1[1] = (int)(row_len << 16);                      // tensor_dim0 lo16
    g1[2] = (int)((row_len >> 16) | (rows << 16));     // dim0 hi16 | dim1 lo16
    g1[3] = (int)((rows >> 16) | (row_len << 16));     // dim1 hi16 | tile_dim0
    g1[4] = (int)rows;                                 // tile_dim1 (tile_dim2=0)
    g1[5] = (int)row_len;                              // tensor_dim0_stride lo32
    g1[6] = 0;
    g1[7] = 0;                                         // dim1_stride unused (2D)
    v4i_ z4 = {0, 0, 0, 0};
    v8i_ z8 = {0, 0, 0, 0, 0, 0, 0, 0};
    __builtin_amdgcn_tensor_load_to_lds(g0, g1, z4, z4, z8, 0);
}
#else
#define HAVE_TDM 0
#endif

// ---------------------------------------------------------------------------
// Kernel 0: build f16 operands in workspace.
//  - Wh: W pre-swizzled into WMMA B-fragment order Wh[tile][kc][lane][i]
//        (fragment = contiguous 32 B per lane -> 2 x global_load_b128 feed)
//  - Uh: plain row-major f16 (TDM raw-copies it into LDS for the scan)
//  - biasv: concat(b_i,b_f,b_c,b_o), folded into xw by kernel 1
// ---------------------------------------------------------------------------
__global__ void __launch_bounds__(256)
convert_wu_kernel(const float* __restrict__ W, const float* __restrict__ U,
                  const float* __restrict__ b_i, const float* __restrict__ b_f,
                  const float* __restrict__ b_c, const float* __restrict__ b_o,
                  _Float16* __restrict__ Wh, _Float16* __restrict__ Uh,
                  float* __restrict__ biasv) {
    int idx = blockIdx.x * 256 + threadIdx.x;
    if (idx < D_ * G_) {
        // destination fragment coords
        int i    = idx & 15;
        int lane = (idx >> 4) & 31;
        int kc   = (idx >> 9) & 3;
        int tile = idx >> 11;
        int K    = kc * 32 + (lane >> 4) * 16 + i;     // B-frag row (K dim)
        int col  = tile * 16 + (lane & 15);            // B-frag col (N dim)
        Wh[idx] = (_Float16)W[(size_t)K * G_ + col];
    } else if (idx < 2 * D_ * G_) {
        int j = idx - D_ * G_;
        Uh[j] = (_Float16)U[j];
    } else if (idx < 2 * D_ * G_ + G_) {
        int g = idx - 2 * D_ * G_;
        float v = (g < 128) ? b_i[g] : (g < 256) ? b_f[g - 128]
                : (g < 384) ? b_c[g - 256] : b_o[g - 384];
        biasv[g] = v;
    }
}

// ---------------------------------------------------------------------------
// Kernel 1: xw = x @ W + bias, written in the scan's C-fragment swizzle:
//   xw[t][tile][lane][e]  (tile = g/16; lane/e = WMMA 16x16 f32 C layout with
//   M = batch b: lane = (b>>3)*16 + (g&15), e = b&7).  One timestep slab is a
//   contiguous 16 KB block, so the scan TDM-loads it and each lane fetches
//   its 8 accumulator values with a single ds_load_b128 per gate tile.
// Grid: 4096 blocks (16 batches x 256 t-tiles), 256 threads (8 waves).
// ---------------------------------------------------------------------------
__global__ void __launch_bounds__(256)
xw_gemm_kernel(const float* __restrict__ x,
               const _Float16* __restrict__ Wh,
               const float* __restrict__ biasv,
               _Float16* __restrict__ xw) {
    __shared__ _Float16 xs[16 * D_];                   // 4 KB staged A tile

    const int tid  = threadIdx.x;
    const int lane = tid & 31;
    const int wave = tid >> 5;
    const int Lhi  = lane >> 4;
    const int Lm   = lane & 15;

    const int blk = blockIdx.x;
    const int b   = blk >> 8;                          // batch
    const int t0  = (blk & 255) * 16;                  // t tile base

    const float* xrow = x + ((size_t)b * T_ + t0) * D_;
#pragma unroll
    for (int i = tid; i < 16 * D_; i += 256) xs[i] = (_Float16)xrow[i];
    __syncthreads();

    v16h afrag[4];
#pragma unroll
    for (int kc = 0; kc < 4; ++kc)
#pragma unroll
        for (int i = 0; i < 16; ++i) {
            int K = kc * 32 + Lhi * 16 + i;
            afrag[kc][i] = xs[Lm * D_ + K];
        }

    // scan-side (lane, e) slot that owns (batch=b, col=tile*16+Lm)
    const int lane_s = ((b >> 3) << 4) + Lm;
    const int e_s    = b & 7;

    const v16h* Whf = (const v16h*)Wh;                 // fragment-order W

#pragma unroll
    for (int q = 0; q < 4; ++q) {
        const int tileIdx = wave * 4 + q;              // col tile 0..31
        const int n0 = tileIdx * 16;
        v8f acc = {};
#pragma unroll
        for (int kc = 0; kc < 4; ++kc) {
            v16h bfrag = Whf[(tileIdx * 4 + kc) * 32 + lane];  // 32 B coalesced
            acc = __builtin_amdgcn_wmma_f32_16x16x32_f16(
                false, afrag[kc], false, bfrag, (short)0, acc, false, false);
        }
        const float bias = biasv[n0 + Lm];
#pragma unroll
        for (int e = 0; e < 8; ++e) {
            int t = t0 + e + 8 * Lhi;                  // M dim here is t
            xw[(((size_t)t * 32 + tileIdx) * 32 + lane_s) * 8 + e_s] =
                (_Float16)(acc[e] + bias);
        }
    }
}

// ---------------------------------------------------------------------------
// Kernel 2: sequential LSTM scan. One persistent workgroup (8 waves).
// U (f16, 128 KB) TDM-loaded into LDS; per-step xw tile (16 KB) TDM
// double-buffered into LDS, overlapped with the step's 16 WMMA per wave.
// ---------------------------------------------------------------------------
__global__ void __launch_bounds__(256)
lstm_scan_kernel(const _Float16* __restrict__ Uh,
                 const _Float16* __restrict__ xw,
                 float* __restrict__ hs, float* __restrict__ cs) {
    extern __shared__ _Float16 smem[];
    _Float16* Us    = smem;                            // [128][512]
    _Float16* hbuf  = smem + D_ * G_;                  // 2 x [16][128]
    _Float16* xwbuf = smem + (LDS_XW_OFF / 2);         // 2 x swizzled 16 KB

    const int tid  = threadIdx.x;
    const int lane = tid & 31;
    const int wave = tid >> 5;
    const int Lhi  = lane >> 4;
    const int Lm   = lane & 15;
    const int col  = wave * 16 + Lm;

#if HAVE_TDM
    if (wave == 0) {
        tdm_load_2d_f16(0u, Uh, D_, G_);                   // U -> LDS
        tdm_load_2d_f16(LDS_XW_OFF, xw, B_, G_);           // xw slab t=0
        __builtin_amdgcn_s_wait_tensorcnt(0);
    }
#else
    for (int i = tid; i < D_ * G_; i += 256) Us[i] = Uh[i];
    for (int i = tid; i < B_ * G_; i += 256) xwbuf[i] = xw[i];
#endif
    for (int i = tid; i < 2 * 16 * H_; i += 256) hbuf[i] = (_Float16)0.0f;
    __syncthreads();

    // Preload B fragments of U: [gate][kchunk] -> 128 VGPRs, live across T loop
    v16h bfr[4][4];
#pragma unroll
    for (int q = 0; q < 4; ++q) {
        const int n0 = q * H_ + wave * 16;
#pragma unroll
        for (int kc = 0; kc < 4; ++kc)
#pragma unroll
            for (int i = 0; i < 16; ++i) {
                int K = kc * 32 + Lhi * 16 + i;
                bfr[q][kc][i] = Us[(size_t)K * G_ + n0 + Lm];
            }
    }

    v8f c = {};                                        // cell state

    for (int t = 0; t < T_; ++t) {
#if HAVE_TDM
        // DMA xw slab for t+1 into the buffer whose readers finished at the
        // previous barrier; overlaps with this step's compute.
        if (wave == 0 && t + 1 < T_)
            tdm_load_2d_f16(LDS_XW_OFF + ((t + 1) & 1) * LDS_XW_TILE,
                            xw + (size_t)(t + 1) * B_ * G_, B_, G_);
#endif
        const _Float16* hb = hbuf + (t & 1) * (16 * H_);
        _Float16*       hn = hbuf + ((t + 1) & 1) * (16 * H_);

        // A fragments: h [16][128] f16 from LDS (2 x ds_load_b128 per chunk)
        v16h af[4];
#pragma unroll
        for (int kc = 0; kc < 4; ++kc)
#pragma unroll
            for (int i = 0; i < 16; ++i) {
                int K = kc * 32 + Lhi * 16 + i;
                af[kc][i] = hb[Lm * H_ + K];
            }

        // accumulators init: swizzled slab -> one ds_load_b128 per gate tile
#if HAVE_TDM
        const v8h* xwt = (const v8h*)(xwbuf + (t & 1) * (B_ * G_));
#else
        const v8h* xwt = (const v8h*)(xw + (size_t)t * B_ * G_);
#endif
        v8f acc[4];
#pragma unroll
        for (int q = 0; q < 4; ++q) {
            const int tileIdx = q * 8 + wave;          // col tile of (gate q, wave)
            v8h xv = xwt[tileIdx * 32 + lane];
#pragma unroll
            for (int e = 0; e < 8; ++e) acc[q][e] = (float)xv[e];
        }

        // z += h @ U : 16 x v_wmma_f32_16x16x32_f16 per wave per step
#pragma unroll
        for (int kc = 0; kc < 4; ++kc)
#pragma unroll
            for (int q = 0; q < 4; ++q)
                acc[q] = __builtin_amdgcn_wmma_f32_16x16x32_f16(
                    false, af[kc], false, bfr[q][kc], (short)0, acc[q], false, false);

        // gates + cell update (wave-local; biases pre-folded into xw)
#pragma unroll
        for (int e = 0; e < 8; ++e) {
            const int bb = e + 8 * Lhi;
            float ig = fast_sigmoid(acc[0][e]);
            float fg = fast_sigmoid(acc[1][e]);
            float gg = fast_tanh   (acc[2][e]);
            float og = fast_sigmoid(acc[3][e]);
            float cn = fg * c[e] + ig * gg;
            float hv = og * fast_tanh(cn);
            c[e] = cn;
            hn[bb * H_ + col] = (_Float16)hv;
            hs[((size_t)bb * T_ + t) * H_ + col] = hv;
            cs[((size_t)bb * T_ + t) * H_ + col] = cn;
        }

#if HAVE_TDM
        if (wave == 0)
            __builtin_amdgcn_s_wait_tensorcnt(0);      // t+1 slab landed
#else
        if (t + 1 < T_)
            for (int i = tid; i < B_ * G_; i += 256)
                xwbuf[i] = xw[(size_t)(t + 1) * B_ * G_ + i];
#endif
        __syncthreads();                               // publish h(t+1) + xw slab
    }
}

extern "C" void kernel_launch(void* const* d_in, const int* in_sizes, int n_in,
                              void* d_out, int out_size, void* d_ws, size_t ws_size,
                              hipStream_t stream) {
    const float* x   = (const float*)d_in[0];
    const float* W   = (const float*)d_in[1];
    const float* U   = (const float*)d_in[2];
    const float* b_i = (const float*)d_in[3];
    const float* b_f = (const float*)d_in[4];
    const float* b_c = (const float*)d_in[5];
    const float* b_o = (const float*)d_in[6];

    float* hs = (float*)d_out;
    float* cs = hs + (size_t)B_ * T_ * H_;

    // workspace layout: xw (64 MB) | Wh (128 KB) | Uh (128 KB) | bias (2 KB)
    _Float16* xw = (_Float16*)d_ws;
    _Float16* Wh = xw + (size_t)T_ * B_ * G_;
    _Float16* Uh = Wh + (size_t)D_ * G_;
    float* biasv = (float*)(Uh + (size_t)D_ * G_);

    convert_wu_kernel<<<dim3(514), dim3(256), 0, stream>>>(
        W, U, b_i, b_f, b_c, b_o, Wh, Uh, biasv);
    xw_gemm_kernel<<<dim3(4096), dim3(256), 0, stream>>>(x, Wh, biasv, xw);

    (void)hipFuncSetAttribute((const void*)lstm_scan_kernel,
                              hipFuncAttributeMaxDynamicSharedMemorySize, LDS_TOTAL);
    lstm_scan_kernel<<<dim3(1), dim3(256), LDS_TOTAL, stream>>>(Uh, xw, hs, cs);
}
